// InfoSelfAttentionLayer_80779744903530
// MI455X (gfx1250) — compile-verified, matrix operand order checked by reading, not run on hardware
//
#include <hip/hip_runtime.h>
#include <hip/hip_bf16.h>

typedef __attribute__((ext_vector_type(16))) __bf16 v16bf;
typedef __attribute__((ext_vector_type(8)))  float  v8f;
typedef __attribute__((ext_vector_type(4)))  unsigned int u32x4;
typedef __attribute__((ext_vector_type(8)))  int i32x8;
typedef __attribute__((ext_vector_type(4)))  int i32x4;

#define D_LLM 768
#define E_OUT 64
#define L_SEQ 2048
#define NSLICE 16            // B*Q
#define NROWS  32768         // B*Q*L

// ---------------------------------------------------------------------------
// Operand gather following CDNA5 16-bit A/B layout (CDNA5 ISA):
// lane holds the non-contraction index r = lane&15; lane half h = lane>>4
// selects K offset +8; VGPR v in 0..3 holds K = h*8 + 2v, +1; VGPR v in 4..7
// holds K = 16 + h*8 + 2(v-4), +1.  Pairs are consecutive -> ds_load_b128.
// ---------------------------------------------------------------------------
__device__ __forceinline__ v16bf load_op(const __bf16* base, int rowStride, int lane) {
    int r  = lane & 15;
    int kb = (lane >> 4) * 8;
    const __bf16* p = base + r * rowStride;
    v16bf out;
#pragma unroll
    for (int v = 0; v < 8; ++v) {
        int k = (v < 4) ? (kb + 2 * v) : (16 + kb + 2 * (v - 4));
        out[2 * v]     = p[k];
        out[2 * v + 1] = p[k + 1];
    }
    return out;
}

__device__ __forceinline__ v8f wmma_bf16(v16bf a, v16bf b, v8f c) {
    return __builtin_amdgcn_wmma_f32_16x16x32_bf16(false, a, false, b,
                                                   (short)0, c, false, false);
}

// ---------------------------------------------------------------------------
// Tensor Data Mover: 2-D tile load, descriptor per CDNA5 ISA
// Group0: count=1 | lds_addr | global_addr[56:0] | type=2.
// Group1: data_size | tensor_dim0/1 (== tile dims, no OOB clip) | tile_dim0/1
//         | tensor_dim0_stride.  Remaining groups zero (<=2-D tensor).
// Tracked by TENSORcnt; pair with __builtin_amdgcn_s_wait_tensorcnt.
// This toolchain exposes the 6-arg builtin (clang-23 / therock-10.0 form).
// ---------------------------------------------------------------------------
__device__ __forceinline__ void tdm_load_2d(void* lds_dst, const void* gsrc,
                                            unsigned dszLog,   // 0=1B 1=2B 2=4B
                                            unsigned tile0,    // elems, dim0
                                            unsigned tile1,    // rows,  dim1
                                            unsigned long long stride0) // elems
{
    unsigned lds_addr = (unsigned)(unsigned long long)(uintptr_t)lds_dst;
    unsigned long long ga = (unsigned long long)(uintptr_t)gsrc;
    u32x4 g0;
    g0[0] = 1u;                                            // count=1, user mode
    g0[1] = lds_addr;                                      // LDS byte address
    g0[2] = (unsigned)(ga & 0xFFFFFFFFu);                  // global addr lo
    g0[3] = (unsigned)((ga >> 32) & 0x1FFFFFFu) | (2u << 30); // addr hi | type=2
    i32x8 g1;
    g1[0] = (int)(dszLog << 16);                           // mask=0 | data_size
    g1[1] = (int)((tile0 & 0xFFFFu) << 16);                // tensor_dim0 lo16
    g1[2] = (int)(((tile0 >> 16) & 0xFFFFu) |              // tensor_dim0 hi16
                  ((tile1 & 0xFFFFu) << 16));              // tensor_dim1 lo16
    g1[3] = (int)(((tile1 >> 16) & 0xFFFFu) |              // tensor_dim1 hi16
                  ((tile0 & 0xFFFFu) << 16));              // tile_dim0
    g1[4] = (int)(tile1 & 0xFFFFu);                        // tile_dim1 | tile_dim2=0
    g1[5] = (int)(stride0 & 0xFFFFFFFFull);                // dim0 stride lo32
    g1[6] = (int)((stride0 >> 32) & 0xFFFFull);            // stride hi16 | d1s lo=0
    g1[7] = 0;
    i32x4 z4 = {0, 0, 0, 0};
    i32x8 z8 = {0, 0, 0, 0, 0, 0, 0, 0};
    __builtin_amdgcn_tensor_load_to_lds(g0, g1, z4, z4, z8, 0);
}

// ---------------------------------------------------------------------------
// Kernel 0: pre-transpose weights to bf16: Wt[mat][n][d], so projection tiles
// become strided 2-D TDM loads.
// ---------------------------------------------------------------------------
__global__ __launch_bounds__(256) void prep_weights(
    const float* __restrict__ Wq, const float* __restrict__ Wk,
    const float* __restrict__ Wv, __bf16* __restrict__ Wt)
{
    const float* Ws[3] = {Wq, Wk, Wv};
    const int total = 3 * E_OUT * D_LLM;
    for (int idx = blockIdx.x * 256 + threadIdx.x; idx < total;
         idx += gridDim.x * 256) {
        int m   = idx / (E_OUT * D_LLM);
        int rem = idx % (E_OUT * D_LLM);
        int n   = rem / D_LLM;
        int d   = rem % D_LLM;
        Wt[idx] = (__bf16)Ws[m][d * E_OUT + n];
    }
}

// ---------------------------------------------------------------------------
// Kernel 1: fused Q/K/V projection.  128 rows per WG (8 waves x 16-row strips),
// d_llm streamed in 32-wide chunks.  Weight tiles arrive via TDM (3 strided
// 2-D descriptors, overlapped with the info f32->bf16 staging).
// ---------------------------------------------------------------------------
__global__ __launch_bounds__(256) void proj_kernel(
    const float* __restrict__ info,
    const float* __restrict__ bq, const float* __restrict__ bk,
    const float* __restrict__ bv, const __bf16* __restrict__ Wt,
    __bf16* __restrict__ Qbf, __bf16* __restrict__ Kbf, __bf16* __restrict__ Vbf)
{
    __shared__ __bf16 sInfo[128][32];                 // 8 KB
    __shared__ alignas(16) __bf16 sW[3][E_OUT][32];   // 12 KB ([n][d] tiles)

    const int tid  = threadIdx.x;
    const int lane = tid & 31;
    const int wave = tid >> 5;
    const long row0 = (long)blockIdx.x * 128;

    v8f zero = {0.f,0.f,0.f,0.f,0.f,0.f,0.f,0.f};
    v8f acc[3][4];
#pragma unroll
    for (int m = 0; m < 3; ++m)
#pragma unroll
        for (int n = 0; n < 4; ++n) acc[m][n] = zero;

    for (int d0 = 0; d0 < D_LLM; d0 += 32) {
        __syncthreads();                 // previous chunk fully consumed
        if (wave == 0) {                 // DMA weight tiles while we stage info
#pragma unroll
            for (int m = 0; m < 3; ++m)
                tdm_load_2d(&sW[m][0][0], Wt + (long)m * E_OUT * D_LLM + d0,
                            /*2B*/1, /*tile0*/32, /*tile1*/E_OUT,
                            /*stride*/D_LLM);
        }
        // info chunk: 128x32 f32 -> bf16
#pragma unroll
        for (int i = 0; i < 16; ++i) {   // 4096/256
            int idx = i * 256 + tid;
            int r = idx >> 5, dd = idx & 31;
            sInfo[r][dd] = (__bf16)info[(row0 + r) * D_LLM + d0 + dd];
        }
        if (wave == 0) __builtin_amdgcn_s_wait_tensorcnt(0);
        __syncthreads();

        v16bf a = load_op(&sInfo[wave * 16][0], 32, lane);
#pragma unroll
        for (int m = 0; m < 3; ++m)
#pragma unroll
            for (int n = 0; n < 4; ++n) {
                v16bf b = load_op(&sW[m][n * 16][0], 32, lane);
                acc[m][n] = wmma_bf16(a, b, acc[m][n]);
            }
    }

    // C/D layout: VGPR v -> row v + (lane>>4)*8, col = lane&15
    const float* bs[3]   = {bq, bk, bv};
    __bf16*      outs[3] = {Qbf, Kbf, Vbf};
    const int h = lane >> 4, col = lane & 15;
#pragma unroll
    for (int m = 0; m < 3; ++m)
#pragma unroll
        for (int n = 0; n < 4; ++n) {
            int   ng   = n * 16 + col;
            float bias = bs[m][ng];
#pragma unroll
            for (int v = 0; v < 8; ++v) {
                long r = row0 + wave * 16 + v + h * 8;
                outs[m][r * E_OUT + ng] = (__bf16)(acc[m][n][v] + bias);
            }
        }
}

// ---------------------------------------------------------------------------
// Kernel 2: attention for one 16-row tile of one slice.
// Phase 1: S = Q Kt via WMMA; K chunks DMA'd by TDM, double-buffered so the
//          next chunk streams in while WMMAs consume the current one.
// Phase 2: two-pass softmax on the full 16x2048 f32 strip in LDS.
// Phase 3: out = P V; V chunk DMA'd by TDM into staging, transposed in LDS.
// ---------------------------------------------------------------------------
__global__ __launch_bounds__(256) void attn_kernel(
    const __bf16* __restrict__ Qbf, const __bf16* __restrict__ Kbf,
    const __bf16* __restrict__ Vbf, float* __restrict__ O)
{
    constexpr int CHUNK = 256;
    constexpr int VTS   = CHUNK + 8;     // padded Vt row stride

    __shared__ float  sS[16][L_SEQ];                     // 128 KB, exp() in place
    __shared__ __bf16 sP[16][L_SEQ];                     // 64 KB
    __shared__ alignas(16) __bf16 sKV[2][E_OUT * VTS];   // 2 x 33 KB
    __shared__ __bf16 sQ[16][E_OUT];                     // 2 KB
    __shared__ float  sRed[16][16];                      // 1 KB
    __shared__ float  sOut[2][16][E_OUT];                // 8 KB

    const int tid  = threadIdx.x;
    const int lane = tid & 31;
    const int wave = tid >> 5;
    const int s    = blockIdx.y;                 // slice (b*Q + q)
    const int r0   = blockIdx.x * 16;            // row tile within slice
    const long base = (long)s * L_SEQ * E_OUT;

    for (int i = tid; i < 16 * E_OUT; i += 256)
        (&sQ[0][0])[i] = Qbf[base + (long)r0 * E_OUT + i];
    __syncthreads();

    const v16bf aQ0 = load_op(&sQ[0][0],  E_OUT, lane);   // e = 0..31
    const v16bf aQ1 = load_op(&sQ[0][32], E_OUT, lane);   // e = 32..63

    // ---- Phase 1: scores, double-buffered TDM K streaming ----
    if (wave == 0)
        tdm_load_2d(&sKV[0][0], Kbf + base, /*4B*/2,
                    CHUNK * E_OUT / 2, 1, CHUNK * E_OUT / 2);
    for (int idx = 0; idx < L_SEQ / CHUNK; ++idx) {
        const int c0 = idx * CHUNK;
        if (wave == 0) __builtin_amdgcn_s_wait_tensorcnt(0);
        __syncthreads();                          // buf[idx&1] ready for all
        if (wave == 0 && c0 + CHUNK < L_SEQ)      // stream next while computing
            tdm_load_2d(&sKV[(idx + 1) & 1][0],
                        Kbf + base + (long)(c0 + CHUNK) * E_OUT, /*4B*/2,
                        CHUNK * E_OUT / 2, 1, CHUNK * E_OUT / 2);
        const __bf16* kb = &sKV[idx & 1][0];
#pragma unroll
        for (int t = 0; t < 2; ++t) {
            int ct = wave * 2 + t;                // 16-col tile within chunk
            v16bf b0 = load_op(kb + (ct * 16) * E_OUT + 0,  E_OUT, lane);
            v16bf b1 = load_op(kb + (ct * 16) * E_OUT + 32, E_OUT, lane);
            v8f c = {0.f,0.f,0.f,0.f,0.f,0.f,0.f,0.f};
            c = wmma_bf16(aQ0, b0, c);
            c = wmma_bf16(aQ1, b1, c);
            int rr = (lane >> 4) * 8, col = lane & 15;
#pragma unroll
            for (int v = 0; v < 8; ++v)
                sS[v + rr][c0 + ct * 16 + col] = c[v];
        }
        __syncthreads();                          // done reading buf[idx&1]
    }

    // ---- Phase 2: softmax(scale * S) over each row, two-pass in LDS ----
    {
        const float scale = 0.125f;               // 1/sqrt(64)
        int r = tid >> 4, sub = tid & 15;
        float mx = -3.402823e38f;
        for (int j = sub; j < L_SEQ; j += 16) mx = fmaxf(mx, sS[r][j]);
        sRed[r][sub] = mx;
        __syncthreads();
        mx = sRed[r][0];
#pragma unroll
        for (int i = 1; i < 16; ++i) mx = fmaxf(mx, sRed[r][i]);

        float sum = 0.f;
        for (int j = sub; j < L_SEQ; j += 16) {
            float e = __expf(scale * (sS[r][j] - mx));
            sS[r][j] = e;
            sum += e;
        }
        __syncthreads();
        sRed[r][sub] = sum;
        __syncthreads();
        sum = 0.f;
#pragma unroll
        for (int i = 0; i < 16; ++i) sum += sRed[r][i];
        float inv = 1.f / sum;
        for (int j = sub; j < L_SEQ; j += 16)
            sP[r][j] = (__bf16)(sS[r][j] * inv);
    }
    __syncthreads();

    // ---- Phase 3: out = P @ V (TDM stage -> LDS transpose -> WMMA) ----
    v8f co = {0.f,0.f,0.f,0.f,0.f,0.f,0.f,0.f};
    const int ntile = wave & 3;                   // 16-wide e tile
    const int half  = wave >> 2;                  // split of the m range
    __bf16* stage = &sKV[0][0];                   // row-major V chunk [m][e]
    __bf16* vt    = &sKV[1][0];                   // transposed, padded stride
    for (int c0 = 0; c0 < L_SEQ; c0 += CHUNK) {
        __syncthreads();                          // prev WMMAs done with vt
        if (wave == 0) {
            tdm_load_2d(stage, Vbf + base + (long)c0 * E_OUT, /*4B*/2,
                        CHUNK * E_OUT / 2, 1, CHUNK * E_OUT / 2);
            __builtin_amdgcn_s_wait_tensorcnt(0);
        }
        __syncthreads();                          // stage ready
#pragma unroll
        for (int e = 0; e < E_OUT; ++e)           // Vt[e][m]: thread owns m=tid
            vt[e * VTS + tid] = stage[tid * E_OUT + e];
        __syncthreads();
#pragma unroll
        for (int ss = 0; ss < 4; ++ss) {
            int m0 = (half * 4 + ss) * 32;        // within chunk
            v16bf a = load_op(&sP[0][c0 + m0], L_SEQ, lane);
            v16bf b = load_op(vt + (ntile * 16) * VTS + m0, VTS, lane);
            co = wmma_bf16(a, b, co);
        }
    }
    {
        int rr = (lane >> 4) * 8, col = lane & 15;
#pragma unroll
        for (int v = 0; v < 8; ++v)
            sOut[half][v + rr][ntile * 16 + col] = co[v];
    }
    __syncthreads();
    for (int i = tid; i < 16 * E_OUT; i += 256) {
        int r = i >> 6, e = i & 63;
        O[base + (long)(r0 + r) * E_OUT + e] = sOut[0][r][e] + sOut[1][r][e];
    }
}

// ---------------------------------------------------------------------------
// Kernel 3: InstanceNorm per slice over (L, E) = 131072 elements.
// ---------------------------------------------------------------------------
__global__ __launch_bounds__(256) void norm_kernel(
    const float* __restrict__ O, float* __restrict__ out)
{
    __shared__ double sSum[256], sSq[256];
    const int  s    = blockIdx.x;
    const long base = (long)s * L_SEQ * E_OUT;
    const int  N    = L_SEQ * E_OUT;

    double sum = 0.0, sq = 0.0;
    for (int i = threadIdx.x; i < N; i += 256) {
        double v = (double)O[base + i];
        sum += v; sq += v * v;
    }
    sSum[threadIdx.x] = sum; sSq[threadIdx.x] = sq;
    __syncthreads();
    for (int off = 128; off > 0; off >>= 1) {
        if (threadIdx.x < off) {
            sSum[threadIdx.x] += sSum[threadIdx.x + off];
            sSq[threadIdx.x]  += sSq[threadIdx.x + off];
        }
        __syncthreads();
    }
    double invN = 1.0 / (double)N;
    double mu   = sSum[0] * invN;
    double var  = sSq[0] * invN - mu * mu;
    float mean = (float)mu;
    float inv  = rsqrtf((float)var + 1e-5f);
    for (int i = threadIdx.x; i < N; i += 256)
        out[base + i] = (O[base + i] - mean) * inv;
}

// ---------------------------------------------------------------------------
extern "C" void kernel_launch(void* const* d_in, const int* in_sizes, int n_in,
                              void* d_out, int out_size, void* d_ws, size_t ws_size,
                              hipStream_t stream)
{
    const float* info = (const float*)d_in[0];
    const float* Wq   = (const float*)d_in[1];
    const float* bq   = (const float*)d_in[2];
    const float* Wk   = (const float*)d_in[3];
    const float* bk   = (const float*)d_in[4];
    const float* Wv   = (const float*)d_in[5];
    const float* bv   = (const float*)d_in[6];

    // workspace: Qbf | Kbf | Vbf (bf16, 4 MB each) | O (f32, 8 MB) | Wt (288 KB)
    __bf16* Qbf = (__bf16*)d_ws;
    __bf16* Kbf = Qbf + (long)NROWS * E_OUT;
    __bf16* Vbf = Kbf + (long)NROWS * E_OUT;
    float*  O   = (float*)(Vbf + (long)NROWS * E_OUT);
    __bf16* Wt  = (__bf16*)(O + (long)NROWS * E_OUT);
    float*  out = (float*)d_out;

    prep_weights<<<32, 256, 0, stream>>>(Wq, Wk, Wv, Wt);
    proj_kernel<<<NROWS / 128, 256, 0, stream>>>(info, bq, bk, bv, Wt,
                                                 Qbf, Kbf, Vbf);
    attn_kernel<<<dim3(L_SEQ / 16, NSLICE), 256, 0, stream>>>(Qbf, Kbf, Vbf, O);
    norm_kernel<<<NSLICE, 256, 0, stream>>>(O, out);
}
